// NeuralSemiLagrangian_85925115723831
// MI455X (gfx1250) — compile-verified, matrix operand order checked by reading, not run on hardware
//
#include <hip/hip_runtime.h>
#include <hip/hip_bf16.h>

typedef _Float16 h16;
typedef __attribute__((ext_vector_type(16))) _Float16 v16h;
typedef __attribute__((ext_vector_type(8)))  float    v8f;
typedef __attribute__((ext_vector_type(4)))  unsigned int u32x4;
typedef __attribute__((ext_vector_type(8)))  int      i32x8;
typedef __attribute__((ext_vector_type(4)))  int      i32x4;

#define BB 2
#define CC 128
#define HH 240
#define WW 480
#define PHH 242
#define PWW 482
#define KTOT 1152   // 9 taps * 128 in-channels
#define NTAPS 9
#define MT 32       // pixels per block (M tile)

#if defined(__gfx1250__) && __has_builtin(__builtin_amdgcn_tensor_load_to_lds)
#define HAVE_TDM 1
#else
#define HAVE_TDM 0
#endif

// ---- geo_pad source-coordinate mapping (pad = 1) ----
__device__ __forceinline__ void pad_src(int ph, int pw, int& h, int& w) {
    int wv = (pw == 0) ? (WW - 1) : ((pw == WW + 1) ? 0 : (pw - 1));
    if (ph == 0)            { h = 0;      w = (wv + WW / 2) % WW; }
    else if (ph == PHH - 1) { h = HH - 1; w = (wv + WW / 2) % WW; }
    else                    { h = ph - 1; w = wv; }
}

// ---- Keys cubic (A = -0.75) ----
__device__ __forceinline__ void cubw(float t, float* wv) {
    const float A = -0.75f;
    float t1 = 1.f + t, t3 = 1.f - t, t4 = 2.f - t;
    wv[0] = ((A * t1 - 5.f * A) * t1 + 8.f * A) * t1 - 4.f * A;
    wv[1] = ((A + 2.f) * t - (A + 3.f)) * t * t + 1.f;
    wv[2] = ((A + 2.f) * t3 - (A + 3.f)) * t3 * t3 + 1.f;
    wv[3] = ((A * t4 - 5.f * A) * t4 + 8.f * A) * t4 - 4.f * A;
}

#if HAVE_TDM
// 2D tile load via the Tensor Data Mover.  D# layout per CDNA5 ISA:
// group0: [1:0]=count=1, [63:32]=lds_addr, [120:64]=global_addr, [127:126]=type=2
// group1: [17:16]=data_size (1 => 2 bytes), [79:48]=tensor_dim0, [111:80]=tensor_dim1,
//         [127:112]=tile_dim0, [143:128]=tile_dim1, [207:160]=tensor_dim0_stride
__device__ __forceinline__ void tdm_load_2d(unsigned lds, unsigned long long ga,
                                            unsigned tile_d0, unsigned tile_d1,
                                            unsigned tensor_d0, unsigned tensor_d1,
                                            unsigned long long stride0) {
    u32x4 g0;
    g0.x = 1u;                                      // count=1, user-mode, no gather
    g0.y = lds;                                     // lds_addr (bytes)
    g0.z = (unsigned)ga;                            // global_addr[31:0]
    g0.w = (unsigned)(ga >> 32) | (2u << 30);       // global_addr[56:32] | type=2
    i32x8 g1;
    g1[0] = (int)(1u << 16);                        // data_size = 2 bytes
    g1[1] = (int)((tensor_d0 & 0xffffu) << 16);     // tensor_dim0[15:0]
    g1[2] = (int)((tensor_d0 >> 16) | ((tensor_d1 & 0xffffu) << 16));
    g1[3] = (int)((tensor_d1 >> 16) | ((tile_d0 & 0xffffu) << 16));
    g1[4] = (int)(tile_d1 & 0xffffu);               // tile_dim1 (tile_dim2 = 0)
    g1[5] = (int)(unsigned)stride0;                 // tensor_dim0_stride[31:0]
    g1[6] = (int)(unsigned)(stride0 >> 32);         // stride[47:32]; dim1_stride = 0
    g1[7] = 0;
    i32x4 z4 = {0, 0, 0, 0};
#if __clang_major__ >= 23
    i32x8 z8 = {0, 0, 0, 0, 0, 0, 0, 0};
    __builtin_amdgcn_tensor_load_to_lds(g0, g1, z4, z4, z8, 0);
#else
    __builtin_amdgcn_tensor_load_to_lds(g0, g1, z4, z4, 0);
#endif
}
#endif

// Build channel-last geo-padded f16 input (B,PH,PW,C); zero LN sums.
__global__ void pad_cast_kernel(const float* __restrict__ x, h16* __restrict__ xp,
                                float* __restrict__ sums) {
    int g = blockIdx.x * 256 + threadIdx.x;
    if (g < 4) sums[g] = 0.f;
    int c = g & (CC - 1);
    int r = g >> 7;
    int pw = r % PWW; r /= PWW;
    int ph = r % PHH; int b = r / PHH;
    int h, w; pad_src(ph, pw, h, w);
    xp[g] = (h16)x[(((size_t)b * CC + c) * HH + h) * WW + w];
}

// OIHW fp32 -> [o][tap*128 + c] f16 (K-major rows for the GEMM B operand)
__global__ void castw_kernel(const float* __restrict__ w, h16* __restrict__ wb) {
    int g = blockIdx.x * 256 + threadIdx.x;   // g = o*1152 + tap*128 + c
    int c = g & 127;
    int r = g >> 7;          // o*9 + tap
    int tap = r % 9;
    int o = r / 9;
    wb[g] = (h16)w[((size_t)(o * CC + c)) * 9 + tap];
}

// Implicit-GEMM 3x3 conv via WMMA f16->f32.
// Block: 256 threads = 8 waves; 32 pixels x 128 out channels.
// Double-buffered TDM pipeline: tap i+1's A(32x128)/B(128x128) tiles stream into
// the other LDS buffer while tap i's 8 WMMAs per wave execute.
template <bool DO_SUMS>
__global__ void conv_wmma_kernel(const h16* __restrict__ Xp, const h16* __restrict__ Wb,
                                 const float* __restrict__ bias, h16* __restrict__ outp,
                                 float* __restrict__ sums, int obase, int Cout) {
    __shared__ __align__(16) h16 A_lds[2][MT][128];    // 2 x 8 KB
    __shared__ __align__(16) h16 B_lds[2][128][128];   // 2 x 32 KB
    __shared__ float red[2];

    const int t = threadIdx.x;
    const int lane = t & 31, wave = t >> 5;
    const int w0 = blockIdx.x * MT;
    const int h  = blockIdx.y;
    const int b  = blockIdx.z;

    if (t == 0) { red[0] = 0.f; red[1] = 0.f; }

    // fragment lane mapping (16-bit A/B wave32 layout):
    // lane&15 = row/col, lane>>4 selects 8-k halves {0..7,16..23} vs {8..15,24..31}
    const int rsel = lane & 15;
    const int kb   = (lane >> 4) << 3;
    const int col  = (wave << 4) + rsel;

    v8f acc0 = {}, acc1 = {};

#if HAVE_TDM
    auto issue_tap = [&](int tap, int buf) {
        const int dy = tap / 3, dx = tap % 3;
        const h16* asrc = Xp + ((size_t)((b * PHH + h + dy) * PWW + (w0 + dx))) * CC;
        tdm_load_2d((unsigned)(uintptr_t)(void*)&A_lds[buf][0][0],
                    (unsigned long long)(uintptr_t)asrc,
                    128u, (unsigned)MT, 128u, (unsigned)MT, (unsigned long long)CC);
        const h16* bsrc = Wb + (size_t)obase * KTOT + tap * 128;
        tdm_load_2d((unsigned)(uintptr_t)(void*)&B_lds[buf][0][0],
                    (unsigned long long)(uintptr_t)bsrc,
                    128u, 128u, (unsigned)KTOT, 128u, (unsigned long long)KTOT);
    };
    if (wave == 0) issue_tap(0, 0);   // prologue: tap 0 into buffer 0
#endif

    for (int tap = 0; tap < NTAPS; ++tap) {
        const int buf = tap & 1;
#if HAVE_TDM
        __syncthreads();   // buffer buf^1 no longer being read by any wave
        if (wave == 0) {   // uniform per-wave branch: one TDM issue stream
            if (tap + 1 < NTAPS) {
                issue_tap(tap + 1, buf ^ 1);
                __builtin_amdgcn_s_wait_tensorcnt(2);  // in-order: tap's pair landed
            } else {
                __builtin_amdgcn_s_wait_tensorcnt(0);
            }
        }
        __syncthreads();   // tap's tiles visible to all waves
#else
        const int dy = tap / 3, dx = tap % 3;
        __syncthreads();
        {   // manual staging fallback: A 32x128 (16 f16/thread), B 128x128 (64 f16/thread)
            const int ap = t >> 3, aj = (t & 7) << 4;
            const h16* s = Xp + ((size_t)((b * PHH + h + dy) * PWW + (w0 + ap + dx))) * CC + aj;
            uint4* a4 = (uint4*)&A_lds[buf][ap][aj];
            a4[0] = ((const uint4*)s)[0];
            a4[1] = ((const uint4*)s)[1];
            const int bo = t >> 1, bj = (t & 1) << 6;
            const h16* s2 = Wb + (size_t)(obase + bo) * KTOT + tap * 128 + bj;
            uint4* d4 = (uint4*)&B_lds[buf][bo][bj];
            const uint4* s4 = (const uint4*)s2;
#pragma unroll
            for (int q = 0; q < 8; ++q) d4[q] = s4[q];
        }
        __syncthreads();
#endif
#pragma unroll
        for (int q = 0; q < 4; ++q) {
            const int kw = q << 5;
            v16h a0, a1, bm;
            ((uint4*)&a0)[0] = *(const uint4*)&A_lds[buf][rsel][kw + kb];
            ((uint4*)&a0)[1] = *(const uint4*)&A_lds[buf][rsel][kw + 16 + kb];
            ((uint4*)&a1)[0] = *(const uint4*)&A_lds[buf][16 + rsel][kw + kb];
            ((uint4*)&a1)[1] = *(const uint4*)&A_lds[buf][16 + rsel][kw + 16 + kb];
            ((uint4*)&bm)[0] = *(const uint4*)&B_lds[buf][col][kw + kb];
            ((uint4*)&bm)[1] = *(const uint4*)&B_lds[buf][col][kw + 16 + kb];
            acc0 = __builtin_amdgcn_wmma_f32_16x16x32_f16(false, a0, false, bm, (short)0, acc0,
                                                          false, false);
            acc1 = __builtin_amdgcn_wmma_f32_16x16x32_f16(false, a1, false, bm, (short)0, acc1,
                                                          false, false);
        }
    }

    // epilogue: D layout -> lane&15 = N (out channel), VGPR r = M row (+8 for hi lanes)
    const int o = obase + (wave << 4) + rsel;
    const float bv = bias[o];
    const int mbase = (lane >> 4) << 3;
    float s1 = 0.f, s2 = 0.f;
#pragma unroll
    for (int r2 = 0; r2 < 8; ++r2) {
        const int m = mbase + r2;
        const float v0 = acc0[r2] + bv;
        const float v1 = acc1[r2] + bv;
        outp[((size_t)((b * HH + h) * WW + w0 + m)) * Cout + o] = (h16)v0;
        outp[((size_t)((b * HH + h) * WW + w0 + 16 + m)) * Cout + o] = (h16)v1;
        if (DO_SUMS) { s1 += v0 + v1; s2 += v0 * v0 + v1 * v1; }
    }
    if (DO_SUMS) {
        atomicAdd(&red[0], s1);
        atomicAdd(&red[1], s2);
        __syncthreads();
        if (t == 0) {
            atomicAdd(&sums[b * 2 + 0], red[0]);
            atomicAdd(&sums[b * 2 + 1], red[1]);
        }
    }
}

// LayerNorm(CHW per batch) + SiLU, writing the geo-padded f16 activation buffer.
__global__ void ln_silu_pad_kernel(const h16* __restrict__ y1, const float* __restrict__ gam,
                                   const float* __restrict__ bet, const float* __restrict__ sums,
                                   h16* __restrict__ apad) {
    int g = blockIdx.x * 256 + threadIdx.x;
    int c = g & 127;
    int r = g >> 7;
    int pw = r % PWW; r /= PWW;
    int ph = r % PHH; int b = r / PHH;
    int h, w; pad_src(ph, pw, h, w);
    const float invN = 1.f / (float)(CC * HH * WW);
    float mean = sums[b * 2] * invN;
    float var  = sums[b * 2 + 1] * invN - mean * mean;
    float rstd = rsqrtf(var + 1e-5f);
    float x = (float)y1[((size_t)((b * HH + h) * WW + w)) * CC + c];
    int sidx = (c * HH + h) * WW + w;
    float v = (x - mean) * rstd * gam[sidx] + bet[sidx];
    v = v / (1.f + __expf(-v));
    apad[g] = (h16)v;
}

// Spherical departure points + 4x4 bicubic gather (lane = channel -> coalesced taps).
__global__ void advect_kernel(const h16* __restrict__ xpad, const h16* __restrict__ vel,
                              const float* __restrict__ latg, const float* __restrict__ long_,
                              const float* __restrict__ dtp, float* __restrict__ out) {
    int g = blockIdx.x * 256 + threadIdx.x;
    int c = g & 127;
    int r = g >> 7;
    int w = r % WW; r /= WW;
    int h = r % HH; int b = r / HH;

    const float dt = *dtp;
    const size_t pix = (size_t)(b * HH + h) * WW + w;
    const float u = (float)vel[pix * 256 + c];
    const float v = (float)vel[pix * 256 + 128 + c];
    const float lat = latg[pix];
    const float lon = long_[pix];

    const float lonp = -u * dt, latp = -v * dt;
    const float slp = __sinf(latp), clp = __cosf(latp);
    const float slo = __sinf(lonp), clo = __cosf(lonp);
    const float sl = __sinf(lat),  cl  = __cosf(lat);
    float sin_lat = slp * cl + clp * clo * sl;
    sin_lat = fminf(fmaxf(sin_lat, -1.f + 1e-7f), 1.f - 1e-7f);
    const float lat_dep = asinf(sin_lat);
    const float num = clp * slo;
    const float den = clp * clo * cl - slp * sl;
    const float TWO_PI = 6.28318530717958647692f;
    float lon_dep = lon + atan2f(num, den);
    lon_dep = fmodf(lon_dep + TWO_PI, TWO_PI);
    if (lon_dep < 0.f) lon_dep += TWO_PI;

    const float gx = (lon_dep * (1.f / TWO_PI) * 2.f - 1.f) * ((float)WW / (float)PWW);
    const float gy = (lat_dep * (1.f / 3.14159265358979f) * 2.f - 1.f) * ((float)HH / (float)PHH);
    float ix = fminf(fmaxf((gx + 1.f) * 0.5f * (float)(PWW - 1), 0.f), (float)(PWW - 1));
    float iy = fminf(fmaxf((gy + 1.f) * 0.5f * (float)(PHH - 1), 0.f), (float)(PHH - 1));
    const float x0f = floorf(ix), y0f = floorf(iy);
    const float tx = ix - x0f, ty = iy - y0f;
    const int x0 = (int)x0f, y0 = (int)y0f;
    float wx[4], wy[4];
    cubw(tx, wx);
    cubw(ty, wy);

    const h16* base = xpad + (size_t)b * PHH * PWW * CC + c;
    float acc = 0.f;
#pragma unroll
    for (int i = 0; i < 4; ++i) {
        const int yi = min(max(y0 + i - 1, 0), PHH - 1);
        float rowv = 0.f;
#pragma unroll
        for (int j = 0; j < 4; ++j) {
            const int xj = min(max(x0 + j - 1, 0), PWW - 1);
            rowv += wx[j] * (float)base[((size_t)yi * PWW + xj) * CC];
        }
        acc += wy[i] * rowv;
    }
    out[(((size_t)b * CC + c) * HH + h) * WW + w] = acc;
}

extern "C" void kernel_launch(void* const* d_in, const int* in_sizes, int n_in,
                              void* d_out, int out_size, void* d_ws, size_t ws_size,
                              hipStream_t stream) {
    const float* hidden = (const float*)d_in[0];
    const float* latg   = (const float*)d_in[1];
    const float* lng    = (const float*)d_in[2];
    const float* w1     = (const float*)d_in[3];
    const float* b1     = (const float*)d_in[4];
    const float* ln_g   = (const float*)d_in[5];
    const float* ln_b   = (const float*)d_in[6];
    const float* w2     = (const float*)d_in[7];
    const float* b2     = (const float*)d_in[8];
    const float* dtp    = (const float*)d_in[9];
    float* out = (float*)d_out;

    // Workspace layout (all 16B-aligned offsets):
    h16* Xpad = (h16*)d_ws;                               // (B,PH,PW,C)  f16
    h16* Wb1  = Xpad + (size_t)BB * PHH * PWW * CC;       // (128,1152)   f16
    h16* Wb2  = Wb1 + (size_t)CC * KTOT;                  // (256,1152)   f16
    h16* y1   = Wb2 + (size_t)256 * KTOT;                 // (B,H,W,128)  f16
    h16* Apad = y1 + (size_t)BB * HH * WW * CC;           // (B,PH,PW,C)  f16
    h16* velb = Apad + (size_t)BB * PHH * PWW * CC;       // (B,H,W,256)  f16
    float* sums = (float*)(velb + (size_t)BB * HH * WW * 256); // 4 floats

    const int padTot = BB * PHH * PWW * CC;               // 29,860,864 (div 256)

    pad_cast_kernel<<<padTot / 256, 256, 0, stream>>>(hidden, Xpad, sums);
    castw_kernel<<<(CC * KTOT) / 256, 256, 0, stream>>>(w1, Wb1);
    castw_kernel<<<(256 * KTOT) / 256, 256, 0, stream>>>(w2, Wb2);

    conv_wmma_kernel<true><<<dim3(WW / MT, HH, BB), 256, 0, stream>>>(
        Xpad, Wb1, b1, y1, sums, 0, CC);

    ln_silu_pad_kernel<<<padTot / 256, 256, 0, stream>>>(y1, ln_g, ln_b, sums, Apad);

    conv_wmma_kernel<false><<<dim3(WW / MT, HH, BB), 256, 0, stream>>>(
        Apad, Wb2, b2, velb, nullptr, 0, 256);
    conv_wmma_kernel<false><<<dim3(WW / MT, HH, BB), 256, 0, stream>>>(
        Apad, Wb2, b2, velb, nullptr, 128, 256);

    advect_kernel<<<(BB * CC * HH * WW) / 256, 256, 0, stream>>>(
        Xpad, velb, latg, lng, dtp, out);
}
